// CMAttention_55714315764288
// MI455X (gfx1250) — matrix operand order repaired and checked by reading, not compiled, not run on hardware
//
#include <hip/hip_runtime.h>
#include <math.h>

typedef _Float16 half_t;
typedef __attribute__((ext_vector_type(16))) _Float16 v16h;
typedef __attribute__((ext_vector_type(8)))  _Float16 v8h;
typedef __attribute__((ext_vector_type(8)))  float    v8f;

#define DIMC   1024
#define HEADS  16
#define DHD    64
#define SEGLEN 1024
#define BATCH  2
#define NSEQ   4096   // NA + NX concatenated sequence per (b)
#define QKN    3072   // 3*DIM

// ---------------------------------------------------------------- utilities

__device__ __forceinline__ v8f zero8(){
  v8f z;
#pragma unroll
  for (int i = 0; i < 8; ++i) z[i] = 0.f;
  return z;
}

__device__ __forceinline__ v8f wmma_f16(v16h a, v16h b, v8f c){
  // D = A(16x32 f16) x B(32x16 f16) + C(16x16 f32)
  return __builtin_amdgcn_wmma_f32_16x16x32_f16(false, a, false, b, (short)0, c,
                                                false, false);
}

// this code's k order for 16-bit A 16x32: lane m = lane&15, half = lane>>4,
// element e -> k = 16*half + e
__device__ __forceinline__ int kpat(int e, int hf){
  return (hf << 4) | e;
}

// A-fragment (16x32 f16): per-lane one 32B run, element e -> k = 16*half + e.
__device__ __forceinline__ v16h load_a_frag(const half_t* base, int row_stride, int lane){
  const int ln = lane & 15, hf = lane >> 4;
  const half_t* p = base + (size_t)ln * row_stride;
  v8h lo = *(const v8h*)(p + 16*hf);
  v8h hi = *(const v8h*)(p + 16*hf + 8);
  v16h r;
#pragma unroll
  for (int i = 0; i < 8; ++i){ r[i] = lo[i]; r[8+i] = hi[i]; }
  return r;
}

// B-fragment (32x16 f16) from [n][k] layout (k contiguous per n):
// lane n = lane&15, element e -> k = 16*half + e  (one 32B run per lane)
__device__ __forceinline__ v16h load_b_frag_kcontig(const half_t* base, int n_stride, int lane){
  const int ln = lane & 15, hf = lane >> 4;
  const half_t* p = base + (size_t)ln * n_stride + 16*hf;
  v8h lo = *(const v8h*)(p);
  v8h hi = *(const v8h*)(p + 8);
  v16h r;
#pragma unroll
  for (int i = 0; i < 8; ++i){ r[i] = lo[i]; r[8+i] = hi[i]; }
  return r;
}

// B-fragment (32x16) from [k][n] layout (k strided): 16 scalar gathers.
__device__ __forceinline__ v16h load_b_frag_kstride(const half_t* base, int k_stride, int lane){
  const int ln = lane & 15, hf = lane >> 4;
  v16h r;
#pragma unroll
  for (int e = 0; e < 16; ++e)
    r[e] = base[(size_t)(16*hf + e) * k_stride + ln];
  return r;
}

// per-lane async copy of 16 bytes global -> LDS (ASYNCcnt-tracked)
__device__ __forceinline__ void async_copy_b128(unsigned lds_off, const void* gptr){
  unsigned long long ga = (unsigned long long)(size_t)gptr;
  asm volatile("global_load_async_to_lds_b128 %0, %1, off"
               :: "v"(lds_off), "v"(ga) : "memory");
}
__device__ __forceinline__ void wait_async0(){
  asm volatile("s_wait_asynccnt 0x0" ::: "memory");
}
// generic shared pointer -> LDS byte offset (low 32 bits of flat LDS address)
__device__ __forceinline__ unsigned lds_offset(const void* p){
  return (unsigned)(size_t)p;
}

// ---------------------------------------------------------------- precision pre-pass
// One-time f32 -> f16 conversion (bandwidth trivial: ~4us at 23.3 TB/s).

__global__ __launch_bounds__(256) void convert_f16(const float* __restrict__ in,
                                                   half_t* __restrict__ out, int n8){
  const int i = blockIdx.x * 256 + threadIdx.x;
  if (i >= n8) return;
  const size_t base = (size_t)i * 8;
  v8h o;
#pragma unroll
  for (int j = 0; j < 8; ++j) o[j] = (half_t)in[base + j];
  *(v8h*)(out + base) = o;
}

// W[1024][3072] f32 -> Wt[3072][1024] f16 (pre-transposed so GEMM B-tiles are [n][k])
__global__ __launch_bounds__(256) void transpose_convert_w(const float* __restrict__ W,
                                                           half_t* __restrict__ Wt){
  const int idx = blockIdx.x * 256 + threadIdx.x;
  if (idx >= DIMC * QKN) return;
  const int n = idx / DIMC, k = idx % DIMC;
  Wt[idx] = (half_t)W[(size_t)k * QKN + n];
}

// ---------------------------------------------------------------- GEMM (qkv projection)
// C[4096,3072] = A[4096,1024] x W[1024,3072]; A and Wt pre-converted f16.
// Block tile 64x128, 8 waves 4(M)x2(N), each wave 16x64 via 4 C-frags.
// Tiles staged with GLOBAL_LOAD_ASYNC_TO_LDS_B128 (no VALU in the staging path).

__global__ __launch_bounds__(256) void gemm_qkv_f16(const half_t* __restrict__ Ah,
                                                    const half_t* __restrict__ Wt,
                                                    half_t* __restrict__ C){
  __shared__ __align__(16) half_t lda[64][32];    // [m][k]
  __shared__ __align__(16) half_t ldbt[128][32];  // [n][k]
  const int tid = threadIdx.x, lane = tid & 31, wave = tid >> 5;
  const int ln = lane & 15, hf = lane >> 4;
  const int m0 = blockIdx.x * 64, n0 = blockIdx.y * 128;
  const int wm = wave & 3, wn = wave >> 2;

  // async staging maps: 16B chunks; A tile = 256 chunks, Wt tile = 512 chunks
  const int arow = tid >> 2, aseg = (tid & 3) * 8;
  const unsigned lda_off  = lds_offset(&lda[arow][aseg]);
  const unsigned ldb_off0 = lds_offset(&ldbt[arow][aseg]);        // rows 0..63
  const unsigned ldb_off1 = lds_offset(&ldbt[64 + arow][aseg]);   // rows 64..127

  v8f acc[4];
#pragma unroll
  for (int nt = 0; nt < 4; ++nt) acc[nt] = zero8();

  for (int k0 = 0; k0 < DIMC; k0 += 32){
    __syncthreads();   // previous iteration's fragment reads are done
    async_copy_b128(lda_off,  Ah + (size_t)(m0 + arow) * DIMC + k0 + aseg);
    async_copy_b128(ldb_off0, Wt + (size_t)(n0 + arow) * DIMC + k0 + aseg);
    async_copy_b128(ldb_off1, Wt + (size_t)(n0 + 64 + arow) * DIMC + k0 + aseg);
    wait_async0();
    __syncthreads();   // all waves' async copies have landed in LDS

    v16h af = load_a_frag(&lda[wm * 16][0], 32, lane);
#pragma unroll
    for (int nt = 0; nt < 4; ++nt){
      v16h bf = load_b_frag_kcontig(&ldbt[wn * 64 + nt * 16][0], 32, lane);
      acc[nt] = wmma_f16(af, bf, acc[nt]);
    }
  }
#pragma unroll
  for (int nt = 0; nt < 4; ++nt)
#pragma unroll
    for (int r = 0; r < 8; ++r)
      C[(size_t)(m0 + wm * 16 + r + 8 * hf) * QKN + n0 + wn * 64 + nt * 16 + ln] =
          (half_t)acc[nt][r];
}

// ---------------------------------------------------------------- norm + rope + pack
// One 64-thread block per (token, head). Packs into [b][h][pos][64] f16.

__global__ __launch_bounds__(64) void norm_rope_pack(const half_t* __restrict__ qkv,
                                                     const float* __restrict__ g_q,
                                                     const float* __restrict__ g_k,
                                                     int posoff,
                                                     half_t* __restrict__ Qh,
                                                     half_t* __restrict__ Kh,
                                                     half_t* __restrict__ Vh){
  const int tok = blockIdx.x;   // b*2048 + n
  const int h   = blockIdx.y;
  const int d   = threadIdx.x;  // 0..63
  const int b = tok >> 11, n = tok & 2047;
  const size_t rb = (size_t)tok * QKN + h * DHD + d;
  float q = (float)qkv[rb];
  float k = (float)qkv[rb + DIMC];
  float v = (float)qkv[rb + 2 * DIMC];

  float sq = q * q, sk = k * k;
#pragma unroll
  for (int m = 1; m < 32; m <<= 1){ sq += __shfl_xor(sq, m, 32); sk += __shfl_xor(sk, m, 32); }
  __shared__ float red[2][2];
  if ((d & 31) == 0){ red[d >> 5][0] = sq; red[d >> 5][1] = sk; }
  __syncthreads();
  const float nq = sqrtf(red[0][0] + red[1][0]);
  const float nk = sqrtf(red[0][1] + red[1][1]);
  q = q / fmaxf(nq, 1e-12f) * g_q[h * DHD + d] * 8.0f;   // * sqrt(64)
  k = k / fmaxf(nk, 1e-12f) * g_k[h * DHD + d] * 8.0f;

  const int pos = posoff + n;
  const float freq = __powf(10000.f, -(float)(d & ~1) / 64.f);
  const float ang = (float)pos * freq;
  const float cs = __cosf(ang), sn = __sinf(ang);
  const float qp = __shfl_xor(q, 1, 32), kp = __shfl_xor(k, 1, 32);
  const float qo = ((d & 1) == 0) ? (q * cs - qp * sn) : (qp * sn + q * cs);
  const float ko = ((d & 1) == 0) ? (k * cs - kp * sn) : (kp * sn + k * cs);

  const size_t oi = ((size_t)(b * HEADS + h) * NSEQ + pos) * DHD + d;
  Qh[oi] = (half_t)qo; Kh[oi] = (half_t)ko; Vh[oi] = (half_t)v;
}

// ---------------------------------------------------------------- infini attention
// One block per (b,h). 8 waves; each wave owns a 16-row query tile per chunk.
// Flash-style online softmax over 32-key tiles; memory matrix M in LDS.

__global__ __launch_bounds__(256) void infini_attention(const half_t* __restrict__ Qg,
                                                        const half_t* __restrict__ Kg,
                                                        const half_t* __restrict__ Vg,
                                                        const float* __restrict__ gate,
                                                        float* __restrict__ AO){
  const int bh = blockIdx.x;          // b*16 + h
  const int hh = bh & (HEADS - 1);
  const half_t* Q = Qg + (size_t)bh * NSEQ * DHD;
  const half_t* K = Kg + (size_t)bh * NSEQ * DHD;
  const half_t* V = Vg + (size_t)bh * NSEQ * DHD;
  float* AOb = AO + (size_t)bh * NSEQ * DHD;
  const float gg = 1.f / (1.f + __expf(-gate[hh]));
  const float sc = 0.125f;            // 64^-0.5
  const int tid = threadIdx.x, lane = tid & 31, wave = tid >> 5;
  const int ln = lane & 15, hf = lane >> 4;

  __shared__ float Ms[64 * 64];                   // f32 memory matrix
  __shared__ __align__(16) half_t Mh[64 * 64];    // f16 mirror for WMMA B-frags
  __shared__ float zs[64];
  __shared__ __align__(16) half_t Pst[8][16][32]; // per-wave P staging (C->A relayout)

  for (int i = tid; i < 64 * 64; i += 256){ Ms[i] = 0.f; Mh[i] = (half_t)0.f; }
  if (tid < 64) zs[tid] = 0.f;
  __syncthreads();

  for (int s = 0; s < 4; ++s){
    const int segbase = s * SEGLEN;

    // ---- local attention + memory read for all 8 query chunks of this segment
    for (int c = 0; c < 8; ++c){
      const int ql0 = c * 128 + wave * 16;   // segment-local first query row
      const int qg0 = segbase + ql0;
      v16h qa0 = load_a_frag(Q + (size_t)qg0 * DHD + 0,  DHD, lane);
      v16h qa1 = load_a_frag(Q + (size_t)qg0 * DHD + 32, DHD, lane);

      v8f o[4];
#pragma unroll
      for (int nt = 0; nt < 4; ++nt) o[nt] = zero8();
      float mrow[8], lrow[8];
#pragma unroll
      for (int r = 0; r < 8; ++r){ mrow[r] = -1e30f; lrow[r] = 0.f; }

      const int kend = ql0 + 16;             // causal key range (segment-local)
      for (int j = 0; j < kend; j += 32){
        if (j + 32 < kend){                  // prefetch next tile (global_prefetch_b8)
          __builtin_prefetch((const void*)(K + (size_t)(segbase + j + 32) * DHD), 0, 1);
          __builtin_prefetch((const void*)(V + (size_t)(segbase + j + 32) * DHD), 0, 1);
        }
        v16h kb00 = load_b_frag_kcontig(K + (size_t)(segbase + j     ) * DHD + 0,  DHD, lane);
        v16h kb01 = load_b_frag_kcontig(K + (size_t)(segbase + j     ) * DHD + 32, DHD, lane);
        v16h kb10 = load_b_frag_kcontig(K + (size_t)(segbase + j + 16) * DHD + 0,  DHD, lane);
        v16h kb11 = load_b_frag_kcontig(K + (size_t)(segbase + j + 16) * DHD + 32, DHD, lane);
        v8f s0 = zero8(), s1 = zero8();
        s0 = wmma_f16(qa0, kb00, s0); s0 = wmma_f16(qa1, kb01, s0);
        s1 = wmma_f16(qa0, kb10, s1); s1 = wmma_f16(qa1, kb11, s1);

        float pscale[8];
#pragma unroll
        for (int r = 0; r < 8; ++r){
          const int qlr = ql0 + r + 8 * hf;
          float v0 = (j + ln      <= qlr) ? s0[r] * sc : -1e30f;
          float v1 = (j + 16 + ln <= qlr) ? s1[r] * sc : -1e30f;
          float tmax = fmaxf(v0, v1);
#pragma unroll
          for (int m = 1; m < 16; m <<= 1) tmax = fmaxf(tmax, __shfl_xor(tmax, m, 32));
          const float mnew = fmaxf(mrow[r], tmax);
          const float ps = __expf(mrow[r] - mnew);
          const float p0 = __expf(v0 - mnew);
          const float p1 = __expf(v1 - mnew);
          float rs = p0 + p1;
#pragma unroll
          for (int m = 1; m < 16; m <<= 1) rs += __shfl_xor(rs, m, 32);
          lrow[r] = lrow[r] * ps + rs;
          mrow[r] = mnew;
          pscale[r] = ps;
          Pst[wave][r + 8 * hf][ln]      = (half_t)p0;
          Pst[wave][r + 8 * hf][16 + ln] = (half_t)p1;
        }
#pragma unroll
        for (int nt = 0; nt < 4; ++nt)
#pragma unroll
          for (int r = 0; r < 8; ++r) o[nt][r] *= pscale[r];

        asm volatile("s_wait_dscnt 0x0" ::: "memory");   // P staging visible to wave
        v16h pa = load_a_frag(&Pst[wave][0][0], 32, lane);
#pragma unroll
        for (int nt = 0; nt < 4; ++nt){
          v16h vb = load_b_frag_kstride(V + (size_t)(segbase + j) * DHD + nt * 16, DHD, lane);
          o[nt] = wmma_f16(pa, vb, o[nt]);
        }
      }

      // memory read: sq = elu(q)+1 is elementwise, so sq A-frag = f(Q A-frag)
      v16h sqa0, sqa1;
#pragma unroll
      for (int e = 0; e < 16; ++e){
        float x0 = (float)qa0[e]; sqa0[e] = (half_t)(x0 > 0.f ? x0 + 1.f : __expf(x0));
        float x1 = (float)qa1[e]; sqa1[e] = (half_t)(x1 > 0.f ? x1 + 1.f : __expf(x1));
      }
      v8f mn[4];
#pragma unroll
      for (int nt = 0; nt < 4; ++nt){
        mn[nt] = zero8();
        v16h mb0 = load_b_frag_kstride(&Mh[0 * 64 + nt * 16],  64, lane);
        v16h mb1 = load_b_frag_kstride(&Mh[32 * 64 + nt * 16], 64, lane);
        mn[nt] = wmma_f16(sqa0, mb0, mn[nt]);
        mn[nt] = wmma_f16(sqa1, mb1, mn[nt]);
      }
      float den = 0.f;
#pragma unroll
      for (int e = 0; e < 16; ++e){
        const int kk = kpat(e, hf);
        den += (float)sqa0[e] * zs[kk] + (float)sqa1[e] * zs[32 + kk];
      }
      den += __shfl_xor(den, 16, 32);     // combine the two k-halves of row ln
      den += 1e-6f;
      float denr[8];
#pragma unroll
      for (int r = 0; r < 8; ++r) denr[r] = __shfl(den, r + 8 * hf, 32);

#pragma unroll
      for (int nt = 0; nt < 4; ++nt)
#pragma unroll
        for (int r = 0; r < 8; ++r){
          const float local = o[nt][r] / lrow[r];
          const float mem   = mn[nt][r] / denr[r];
          AOb[(size_t)(qg0 + r + 8 * hf) * DHD + nt * 16 + ln] =
              gg * mem + (1.f - gg) * local;
        }
    }
    __syncthreads();

    // ---- M += sk^T @ v over this segment (K-dim = 1024). 16 tiles, 2 per wave.
#pragma unroll
    for (int tt = 0; tt < 2; ++tt){
      const int t = wave * 2 + tt;
      const int dt = t >> 2, et = t & 3;
      v8f acc = zero8();
      for (int l = 0; l < SEGLEN; l += 32){
        v16h af;   // A[m=d][k=l] = elu(K[l][d])+1  (transposed gather)
#pragma unroll
        for (int e = 0; e < 16; ++e){
          const int li = l + kpat(e, hf);
          const float xv = (float)K[(size_t)(segbase + li) * DHD + dt * 16 + ln];
          af[e] = (half_t)(xv > 0.f ? xv + 1.f : __expf(xv));
        }
        v16h bf;   // B[k=l][n=e-col] = V[l][e]
#pragma unroll
        for (int e = 0; e < 16; ++e)
          bf[e] = V[(size_t)(segbase + l + 16 * hf + e) * DHD + et * 16 + ln];
        acc = wmma_f16(af, bf, acc);
      }
#pragma unroll
      for (int r = 0; r < 8; ++r)
        Ms[(dt * 16 + r + 8 * hf) * 64 + et * 16 + ln] += acc[r];
    }
    if (tid < 64){                       // z += sum_l (elu(k)+1)
      float zacc = 0.f;
      for (int l = 0; l < SEGLEN; ++l){
        const float xv = (float)K[(size_t)(segbase + l) * DHD + tid];
        zacc += (xv > 0.f ? xv + 1.f : __expf(xv));
      }
      zs[tid] += zacc;
    }
    __syncthreads();
    for (int i = tid; i < 64 * 64; i += 256) Mh[i] = (half_t)Ms[i];
    __syncthreads();
  }
}

// ---------------------------------------------------------------- final layout merge

__global__ __launch_bounds__(256) void merge_out(const float* __restrict__ AO,
                                                 float* __restrict__ out){
  const size_t HALF = (size_t)BATCH * 2048 * DIMC;   // out_x element count
  const size_t idx = (size_t)blockIdx.x * 256 + threadIdx.x;
  if (idx >= 2 * HALF) return;
  const int isA = idx >= HALF;
  const size_t jx = isA ? idx - HALF : idx;
  const int b   = (int)(jx / (2048 * DIMC));
  const int rem = (int)(jx % (2048 * DIMC));
  const int t   = rem / DIMC;
  const int col = rem % DIMC;
  const int h = col >> 6, d = col & 63;
  const int pos = isA ? t : (2048 + t);
  out[idx] = AO[((size_t)(b * HEADS + h) * NSEQ + pos) * DHD + d];
}

// ---------------------------------------------------------------- launcher

extern "C" void kernel_launch(void* const* d_in, const int* in_sizes, int n_in,
                              void* d_out, int out_size, void* d_ws, size_t ws_size,
                              hipStream_t stream){
  (void)in_sizes; (void)n_in; (void)out_size; (void)ws_size;
  const float* x    = (const float*)d_in[0];
  const float* a    = (const float*)d_in[1];
  const float* Wx   = (const float*)d_in[2];
  const float* Wa   = (const float*)d_in[3];
  const float* g_qx = (const float*)d_in[4];
  const float* g_kx = (const float*)d_in[5];
  const float* g_qa = (const float*)d_in[6];
  const float* g_ka = (const float*)d_in[7];
  const float* gate = (const float*)d_in[8];

  char* ws = (char*)d_ws;
  size_t off = 0;
  auto take = [&](size_t bytes) -> void* {
    void* p = ws + off;
    off += (bytes + 255) & ~(size_t)255;
    return p;
  };
  const size_t NTOK = (size_t)BATCH * 2048;                 // tokens per source
  half_t* xh   = (half_t*)take(NTOK * DIMC * sizeof(half_t));
  half_t* ah   = (half_t*)take(NTOK * DIMC * sizeof(half_t));
  half_t* Wxt  = (half_t*)take((size_t)DIMC * QKN * sizeof(half_t));
  half_t* Wat  = (half_t*)take((size_t)DIMC * QKN * sizeof(half_t));
  half_t* qkv_a = (half_t*)take(NTOK * QKN * sizeof(half_t));
  half_t* qkv_x = (half_t*)take(NTOK * QKN * sizeof(half_t));
  const size_t qkvsz = (size_t)BATCH * HEADS * NSEQ * DHD;
  half_t* Qh = (half_t*)take(qkvsz * sizeof(half_t));
  half_t* Kh = (half_t*)take(qkvsz * sizeof(half_t));
  half_t* Vh = (half_t*)take(qkvsz * sizeof(half_t));
  float*  AO = (float*)take(qkvsz * sizeof(float));

  // precision pre-pass (f32 -> f16; weights also transposed for [n][k] staging)
  const int n8 = (int)(NTOK * DIMC / 8);
  convert_f16<<<(n8 + 255) / 256, 256, 0, stream>>>(x, xh, n8);
  convert_f16<<<(n8 + 255) / 256, 256, 0, stream>>>(a, ah, n8);
  const int wtot = DIMC * QKN;
  transpose_convert_w<<<(wtot + 255) / 256, 256, 0, stream>>>(Wx, Wxt);
  transpose_convert_w<<<(wtot + 255) / 256, 256, 0, stream>>>(Wa, Wat);

  dim3 gg(64, 24);   // 4096/64 x 3072/128
  gemm_qkv_f16<<<gg, 256, 0, stream>>>(ah, Wat, qkv_a);
  gemm_qkv_f16<<<gg, 256, 0, stream>>>(xh, Wxt, qkv_x);

  dim3 gn((unsigned)NTOK, HEADS);
  norm_rope_pack<<<gn, 64, 0, stream>>>(qkv_a, g_qa, g_ka, 0,    Qh, Kh, Vh);
  norm_rope_pack<<<gn, 64, 0, stream>>>(qkv_x, g_qx, g_kx, 2048, Qh, Kh, Vh);

  infini_attention<<<BATCH * HEADS, 256, 0, stream>>>(Qh, Kh, Vh, gate, AO);

  const size_t total = 2 * (size_t)BATCH * 2048 * DIMC;
  merge_out<<<(unsigned)((total + 255) / 256), 256, 0, stream>>>(AO, (float*)d_out);
}
